// PostNormAttentionBlock_55233279426864
// MI455X (gfx1250) — compile-verified, matrix operand order checked by reading, not run on hardware
//
#include <hip/hip_runtime.h>
#include <math.h>

typedef __attribute__((ext_vector_type(16))) __bf16        v16bf;
typedef __attribute__((ext_vector_type(8)))  float         v8f;
typedef __attribute__((ext_vector_type(4)))  unsigned int  uintx4;
typedef int v4i __attribute__((__vector_size__(4 * sizeof(int))));

#define BB   2
#define SS   2048
#define DD   1024
#define HH   16
#define DHh  64
#define FF_  4096
#define MTOK (BB*SS)   // 4096 tokens

#if defined(__has_builtin)
#if __has_builtin(__builtin_amdgcn_global_load_async_to_lds_b128)
#define HAVE_ASYNC 1
#endif
#endif
#ifndef HAVE_ASYNC
#define HAVE_ASYNC 0
#endif

__device__ __forceinline__ unsigned short f2bf(float f) {
    unsigned u = __float_as_uint(f);
    u += 0x7fffu + ((u >> 16) & 1u);          // round-to-nearest-even
    return (unsigned short)(u >> 16);
}

union OpAB {          // one WMMA 16x32 bf16 operand (32B per lane)
    v16bf  v;
    uintx4 q[2];
};

#if HAVE_ASYNC
// async global(16B/lane) -> LDS, tracked by ASYNCcnt.  AS pointers built via
// integer casts: flat low 32 bits are the LDS offset (ISA aperture rule);
// flat address of global memory equals the global address.
__device__ __forceinline__ void async_b128(const unsigned short* g, unsigned short* l) {
    __builtin_amdgcn_global_load_async_to_lds_b128(
        (__attribute__((address_space(1))) v4i*)(unsigned long long)(size_t)g,
        (__attribute__((address_space(3))) v4i*)(unsigned)(size_t)l,
        0, 0);
}
template<int N>
__device__ __forceinline__ void wait_async() {
#if __has_builtin(__builtin_amdgcn_s_wait_asynccnt)
    __builtin_amdgcn_s_wait_asynccnt((unsigned short)N);
#else
    asm volatile("s_wait_asynccnt %0" :: "i"(N) : "memory");
#endif
}
#endif

// ---------------- elementwise fp32 -> bf16 cast ----------------
__global__ void k_cast_bf16(const float* __restrict__ in,
                            unsigned short* __restrict__ out, int n) {
    int i = blockIdx.x * 256 + threadIdx.x;
    if (i < n) out[i] = f2bf(in[i]);
}

// ---------------- fp32 [K][N] -> bf16 W^T [N][K] ----------------
__global__ void k_transpose_bf16(const float* __restrict__ in,
                                 unsigned short* __restrict__ out, int K, int N) {
    int i = blockIdx.x * 256 + threadIdx.x;
    if (i < K * N) {
        int k = i / N, n = i % N;
        out[(size_t)n * K + k] = f2bf(in[i]);
    }
}

// one 32-deep K step of WMMA on a staged 128x32 A tile and 128x32 B tile
__device__ __forceinline__ void wmma_tiles(const unsigned short* __restrict__ As,
                                           const unsigned short* __restrict__ Bs,
                                           v8f (&acc)[4][2],
                                           int wm, int wn, int half, int l16)
{
    OpAB aop[4], bop[2];
#pragma unroll
    for (int i = 0; i < 4; ++i) {
        int r = (wm * 64 + i * 16 + l16) * 32;
        aop[i].q[0] = *(const uintx4*)&As[r + half * 8];
        aop[i].q[1] = *(const uintx4*)&As[r + 16 + half * 8];
    }
#pragma unroll
    for (int j = 0; j < 2; ++j) {
        int r = (wn * 32 + j * 16 + l16) * 32;
        bop[j].q[0] = *(const uintx4*)&Bs[r + half * 8];
        bop[j].q[1] = *(const uintx4*)&Bs[r + 16 + half * 8];
    }
#pragma unroll
    for (int i = 0; i < 4; ++i)
#pragma unroll
        for (int j = 0; j < 2; ++j)
            acc[i][j] = __builtin_amdgcn_wmma_f32_16x16x32_bf16(
                false, aop[i].v, false, bop[j].v, (short)0, acc[i][j],
                false, false);
}

// ---------------- bf16 WMMA GEMM: C[M][N] = A[M][K] * BT[N][K]^T + bias ----
// MODE 0: f32 flat   MODE 1: bf16 [b,h,s,dh]   MODE 2: bf16 [b,h,dh,s] (V^T)
// MODE 3: GELU -> bf16 flat
template<int MODE>
__global__ __launch_bounds__(256)
void k_gemm(const unsigned short* __restrict__ A,
            const unsigned short* __restrict__ BT,
            const float* __restrict__ bias, void* __restrict__ out,
            int M, int N, int K)
{
    const int tid  = threadIdx.x;
    const int lane = tid & 31, wid = tid >> 5;
    const int wm   = wid & 1,  wn  = wid >> 1;       // 2x4 wave grid
    const int half = lane >> 4, l16 = lane & 15;
    const int m0   = blockIdx.y * 128;
    const int n0   = blockIdx.x * 128;

    v8f acc[4][2] = {};

#if HAVE_ASYNC
    // double-buffered async global->LDS pipeline (4 async b128 per thread/step)
    __shared__ __align__(16) unsigned short As[2][128 * 32];
    __shared__ __align__(16) unsigned short Bs[2][128 * 32];

    const int nk = K >> 5;
    // preload k-step 0 into buffer 0
#pragma unroll
    for (int c = 0; c < 2; ++c) {
        int ch = tid + c * 256, row = ch >> 2, ko = (ch & 3) * 8;
        async_b128(A  + (size_t)(m0 + row) * K + ko, &As[0][row * 32 + ko]);
        async_b128(BT + (size_t)(n0 + row) * K + ko, &Bs[0][row * 32 + ko]);
    }
    for (int kt = 0; kt < nk; ++kt) {
        const int cur = kt & 1;
        if (kt + 1 < nk) {
            const int k1 = (kt + 1) << 5;
#pragma unroll
            for (int c = 0; c < 2; ++c) {
                int ch = tid + c * 256, row = ch >> 2, ko = (ch & 3) * 8;
                async_b128(A  + (size_t)(m0 + row) * K + k1 + ko,
                           &As[cur ^ 1][row * 32 + ko]);
                async_b128(BT + (size_t)(n0 + row) * K + k1 + ko,
                           &Bs[cur ^ 1][row * 32 + ko]);
            }
            wait_async<4>();    // in-order completion => current buffer done
        } else {
            wait_async<0>();
        }
        __syncthreads();        // make all waves' async data visible
        wmma_tiles(As[cur], Bs[cur], acc, wm, wn, half, l16);
        __syncthreads();        // finish reads before buffer is overwritten
    }
#else
    // fallback: global->VGPR->LDS staging
    __shared__ __align__(16) unsigned short As[128 * 32];
    __shared__ __align__(16) unsigned short Bs[128 * 32];

    for (int k0 = 0; k0 < K; k0 += 32) {
        uintx4 av[2], bv[2];
#pragma unroll
        for (int c = 0; c < 2; ++c) {
            int ch = tid + c * 256, row = ch >> 2, ko = (ch & 3) * 8;
            av[c] = *(const uintx4*)(A  + (size_t)(m0 + row) * K + k0 + ko);
            bv[c] = *(const uintx4*)(BT + (size_t)(n0 + row) * K + k0 + ko);
        }
        __syncthreads();
#pragma unroll
        for (int c = 0; c < 2; ++c) {
            int ch = tid + c * 256, row = ch >> 2, ko = (ch & 3) * 8;
            *(uintx4*)&As[row * 32 + ko] = av[c];
            *(uintx4*)&Bs[row * 32 + ko] = bv[c];
        }
        __syncthreads();
        wmma_tiles(As, Bs, acc, wm, wn, half, l16);
    }
#endif

#pragma unroll
    for (int j = 0; j < 2; ++j) {
        int   gN = n0 + wn * 32 + j * 16 + l16;
        float bb = bias[gN];
#pragma unroll
        for (int i = 0; i < 4; ++i) {
#pragma unroll
            for (int r = 0; r < 8; ++r) {
                int   gM = m0 + wm * 64 + i * 16 + r + half * 8;
                float v  = acc[i][j][r] + bb;
                if (MODE == 0) {
                    ((float*)out)[(size_t)gM * N + gN] = v;
                } else if (MODE == 1) {           // bf16 [b,h,s,dh]
                    int b = gM / SS, s = gM % SS;
                    int h = gN >> 6, dh = gN & 63;
                    ((unsigned short*)out)[(((size_t)b * HH + h) * SS + s) * DHh + dh] = f2bf(v);
                } else if (MODE == 2) {           // bf16 [b,h,dh,s]
                    int b = gM / SS, s = gM % SS;
                    int h = gN >> 6, dh = gN & 63;
                    ((unsigned short*)out)[(((size_t)b * HH + h) * DHh + dh) * SS + s] = f2bf(v);
                } else {                          // GELU -> bf16 flat
                    float g = 0.5f * v * (1.0f + erff(v * 0.70710678f));
                    ((unsigned short*)out)[(size_t)gM * N + gN] = f2bf(g);
                }
            }
        }
    }
}

// ---------------- flash attention: per (b,h), 64 queries / block -----------
__global__ __launch_bounds__(128)
void k_attention(const unsigned short* __restrict__ Q,
                 const unsigned short* __restrict__ Kmat,
                 const unsigned short* __restrict__ VT,
                 const int* __restrict__ mask,
                 unsigned short* __restrict__ out)
{
    __shared__ __align__(16) unsigned short Pls[4][16 * 32];

    const int tid  = threadIdx.x;
    const int lane = tid & 31, wid = tid >> 5;
    const int half = lane >> 4, l16 = lane & 15;
    const int qbase = blockIdx.x * 64 + wid * 16;
    const int bh = blockIdx.y;
    const int b  = bh / HH, h = bh % HH;
    const int* mrow = mask + b * SS;

    // Q operands (A layout): 16 rows x dh[0..31] and dh[32..63]
    const unsigned short* Qp = Q + ((size_t)bh * SS + qbase + l16) * DHh;
    OpAB q0, q1;
    q0.q[0] = *(const uintx4*)(Qp + half * 8);
    q0.q[1] = *(const uintx4*)(Qp + 16 + half * 8);
    q1.q[0] = *(const uintx4*)(Qp + 32 + half * 8);
    q1.q[1] = *(const uintx4*)(Qp + 48 + half * 8);

    v8f   O[4] = {};
    float mrun[8], lrun[8];
#pragma unroll
    for (int r = 0; r < 8; ++r) { mrun[r] = -INFINITY; lrun[r] = 0.f; }

    for (int kb = 0; kb < SS; kb += 32) {
        // prefetch next key-block (K rows and V^T rows) into caches
        if (kb + 32 < SS) {
            __builtin_prefetch(Kmat + ((size_t)bh * SS + kb + 32 + l16) * DHh, 0, 3);
            __builtin_prefetch(VT + ((size_t)bh * DHh + l16) * SS + kb + 32, 0, 3);
        }
        v8f sc[2];
#pragma unroll
        for (int t = 0; t < 2; ++t) {             // two 16-key score tiles
            int key = kb + t * 16 + l16;
            const unsigned short* Kp = Kmat + ((size_t)bh * SS + key) * DHh;
            OpAB kb0, kb1;
            kb0.q[0] = *(const uintx4*)(Kp + half * 8);
            kb0.q[1] = *(const uintx4*)(Kp + 16 + half * 8);
            kb1.q[0] = *(const uintx4*)(Kp + 32 + half * 8);
            kb1.q[1] = *(const uintx4*)(Kp + 48 + half * 8);
            v8f z = {};
            v8f c = __builtin_amdgcn_wmma_f32_16x16x32_bf16(
                false, q0.v, false, kb0.v, (short)0, z, false, false);
            c = __builtin_amdgcn_wmma_f32_16x16x32_bf16(
                false, q1.v, false, kb1.v, (short)0, c, false, false);
            int mv = mrow[key];
#pragma unroll
            for (int r = 0; r < 8; ++r)
                c[r] = (mv == 0) ? -INFINITY : c[r] * 0.125f;  // 1/sqrt(64)
            sc[t] = c;
        }

        // streaming softmax: row = (r + half*8); 16 lanes share a row
#pragma unroll
        for (int r = 0; r < 8; ++r) {
            float rm = fmaxf(sc[0][r], sc[1][r]);
#pragma unroll
            for (int off = 8; off >= 1; off >>= 1)
                rm = fmaxf(rm, __shfl_xor(rm, off, 16));
            float mnew  = fmaxf(fmaxf(mrun[r], rm), -1e30f);
            float alpha = __expf(mrun[r] - mnew);
            float p0 = __expf(sc[0][r] - mnew);
            float p1 = __expf(sc[1][r] - mnew);
            sc[0][r] = p0; sc[1][r] = p1;
            float rs = p0 + p1;
#pragma unroll
            for (int off = 8; off >= 1; off >>= 1)
                rs += __shfl_xor(rs, off, 16);
            lrun[r] = lrun[r] * alpha + rs;
            mrun[r] = mnew;
            O[0][r] *= alpha; O[1][r] *= alpha; O[2][r] *= alpha; O[3][r] *= alpha;
        }

        // P: C-layout f32 -> A-layout bf16 via per-wave LDS tile
        unsigned short* Pw = Pls[wid];
#pragma unroll
        for (int t = 0; t < 2; ++t)
#pragma unroll
            for (int r = 0; r < 8; ++r)
                Pw[(r + half * 8) * 32 + t * 16 + l16] = f2bf(sc[t][r]);
        __syncthreads();
        OpAB pa;
        pa.q[0] = *(const uintx4*)&Pw[l16 * 32 + half * 8];
        pa.q[1] = *(const uintx4*)&Pw[l16 * 32 + 16 + half * 8];

        // O += P[16x32] @ V[32 x 64], V^T layout gives contiguous B operands
#pragma unroll
        for (int t = 0; t < 4; ++t) {
            int dh = t * 16 + l16;
            const unsigned short* Vp = VT + ((size_t)bh * DHh + dh) * SS + kb;
            OpAB vb;
            vb.q[0] = *(const uintx4*)(Vp + half * 8);
            vb.q[1] = *(const uintx4*)(Vp + 16 + half * 8);
            O[t] = __builtin_amdgcn_wmma_f32_16x16x32_bf16(
                false, pa.v, false, vb.v, (short)0, O[t], false, false);
        }
        __syncthreads();
    }

    // normalize and emit merged-head bf16 [b,s,D]
#pragma unroll
    for (int t = 0; t < 4; ++t)
#pragma unroll
        for (int r = 0; r < 8; ++r) {
            int s = qbase + r + half * 8;
            int c = h * DHh + t * 16 + l16;
            out[((size_t)b * SS + s) * DD + c] = f2bf(O[t][r] / lrun[r]);
        }
}

// ---------------- fused residual + LayerNorm (f32 out, optional bf16) -----
__global__ __launch_bounds__(256)
void k_ln_residual(const float* __restrict__ a, const float* __restrict__ res,
                   const float* __restrict__ w, const float* __restrict__ bi,
                   float* __restrict__ outF, unsigned short* __restrict__ outB)
{
    __shared__ float red1[256], red2[256];
    const int    tid  = threadIdx.x;
    const size_t base = (size_t)blockIdx.x * DD;
    float s = 0.f, s2 = 0.f;
    for (int i = tid; i < DD; i += 256) {
        float v = a[base + i] + res[base + i];
        s += v; s2 += v * v;
    }
    red1[tid] = s; red2[tid] = s2; __syncthreads();
    for (int st = 128; st > 0; st >>= 1) {
        if (tid < st) { red1[tid] += red1[tid + st]; red2[tid] += red2[tid + st]; }
        __syncthreads();
    }
    float mean = red1[0] * (1.0f / DD);
    float var  = red2[0] * (1.0f / DD) - mean * mean;
    float rstd = rsqrtf(var + 1e-5f);
    for (int i = tid; i < DD; i += 256) {
        float v = (a[base + i] + res[base + i] - mean) * rstd * w[i] + bi[i];
        outF[base + i] = v;
        if (outB) outB[base + i] = f2bf(v);
    }
}

extern "C" void kernel_launch(void* const* d_in, const int* in_sizes, int n_in,
                              void* d_out, int out_size, void* d_ws, size_t ws_size,
                              hipStream_t stream)
{
    (void)in_sizes; (void)n_in; (void)out_size; (void)ws_size;
    const float* x    = (const float*)d_in[0];
    const int*   mask = (const int*)  d_in[1];
    const float* WQ = (const float*)d_in[2];  const float* bQ = (const float*)d_in[3];
    const float* WK = (const float*)d_in[4];  const float* bK = (const float*)d_in[5];
    const float* WV = (const float*)d_in[6];  const float* bV = (const float*)d_in[7];
    const float* WY = (const float*)d_in[8];  const float* bY = (const float*)d_in[9];
    const float* ln1w = (const float*)d_in[10]; const float* ln1b = (const float*)d_in[11];
    const float* ln2w = (const float*)d_in[12]; const float* ln2b = (const float*)d_in[13];
    const float* W1 = (const float*)d_in[14]; const float* b1 = (const float*)d_in[15];
    const float* W2 = (const float*)d_in[16]; const float* b2 = (const float*)d_in[17];

    char* p = (char*)d_ws;
    auto alloc = [&](size_t bytes) {
        char* r = p; p += (bytes + 255) & ~(size_t)255; return (void*)r;
    };
    unsigned short* xb    = (unsigned short*)alloc((size_t)MTOK * DD * 2);
    unsigned short* wqT   = (unsigned short*)alloc((size_t)DD * DD * 2);
    unsigned short* wkT   = (unsigned short*)alloc((size_t)DD * DD * 2);
    unsigned short* wvT   = (unsigned short*)alloc((size_t)DD * DD * 2);
    unsigned short* wyT   = (unsigned short*)alloc((size_t)DD * DD * 2);
    unsigned short* w1T   = (unsigned short*)alloc((size_t)DD * FF_ * 2);
    unsigned short* w2T   = (unsigned short*)alloc((size_t)FF_ * DD * 2);
    unsigned short* Qb    = (unsigned short*)alloc((size_t)MTOK * DD * 2);
    unsigned short* Kb    = (unsigned short*)alloc((size_t)MTOK * DD * 2);
    unsigned short* VTb   = (unsigned short*)alloc((size_t)MTOK * DD * 2);
    unsigned short* attnb = (unsigned short*)alloc((size_t)MTOK * DD * 2);
    float*          proj  = (float*)alloc((size_t)MTOK * DD * 4);
    float*          h1f   = (float*)alloc((size_t)MTOK * DD * 4);
    unsigned short* h1b   = (unsigned short*)alloc((size_t)MTOK * DD * 2);
    unsigned short* gel   = (unsigned short*)alloc((size_t)MTOK * FF_ * 2);
    float*          ff2   = (float*)alloc((size_t)MTOK * DD * 4);

    int n = MTOK * DD;
    k_cast_bf16<<<n / 256, 256, 0, stream>>>(x, xb, n);
    n = DD * DD;
    k_transpose_bf16<<<n / 256, 256, 0, stream>>>(WQ, wqT, DD, DD);
    k_transpose_bf16<<<n / 256, 256, 0, stream>>>(WK, wkT, DD, DD);
    k_transpose_bf16<<<n / 256, 256, 0, stream>>>(WV, wvT, DD, DD);
    k_transpose_bf16<<<n / 256, 256, 0, stream>>>(WY, wyT, DD, DD);
    n = DD * FF_;
    k_transpose_bf16<<<n / 256, 256, 0, stream>>>(W1, w1T, DD, FF_);
    k_transpose_bf16<<<n / 256, 256, 0, stream>>>(W2, w2T, FF_, DD);

    dim3 g1(DD / 128, MTOK / 128);       // (8, 32)
    k_gemm<1><<<g1, 256, 0, stream>>>(xb, wqT, bQ, Qb,  MTOK, DD, DD);
    k_gemm<1><<<g1, 256, 0, stream>>>(xb, wkT, bK, Kb,  MTOK, DD, DD);
    k_gemm<2><<<g1, 256, 0, stream>>>(xb, wvT, bV, VTb, MTOK, DD, DD);

    dim3 ga(SS / 64, BB * HH);           // (32, 32)
    k_attention<<<ga, 128, 0, stream>>>(Qb, Kb, VTb, mask, attnb);

    k_gemm<0><<<g1, 256, 0, stream>>>(attnb, wyT, bY, proj, MTOK, DD, DD);
    k_ln_residual<<<MTOK, 256, 0, stream>>>(proj, x, ln1w, ln1b, h1f, h1b);

    dim3 g2(FF_ / 128, MTOK / 128);      // (32, 32)
    k_gemm<3><<<g2, 256, 0, stream>>>(h1b, w1T, b1, gel, MTOK, FF_, DD);
    k_gemm<0><<<g1, 256, 0, stream>>>(gel, w2T, b2, ff2, MTOK, DD, FF_);
    k_ln_residual<<<MTOK, 256, 0, stream>>>(ff2, h1f, ln2w, ln2b, (float*)d_out, nullptr);
}
